// UpBlock_63591285785185
// MI455X (gfx1250) — matrix operand order repaired and checked by reading, not compiled, not run on hardware
//
#include <hip/hip_runtime.h>

// ---------------------------------------------------------------------------
// Sparse-conv UpBlock for MI455X (gfx1250, wave32, WMMA).
// Gather -> bf16 WMMA GEMM (f32 accum) -> f32 atomic scatter, per kernel tap.
// BN in training mode via per-channel atomic reduction.
// ---------------------------------------------------------------------------

#define C       128
#define TILE_P  64
#define LDA     136   // padded bf16 row stride for gathered A tile
#define LDB     136   // padded bf16 row stride for weights (stored [n][k])

typedef __attribute__((ext_vector_type(16))) __bf16 v16bf;
typedef __attribute__((ext_vector_type(8)))  float  v8f;

union FragAB { v16bf v; unsigned short u[16]; };

__device__ __forceinline__ unsigned short f2bf(float x) {
  union { float f; unsigned int i; } c; c.f = x;
  unsigned int u = c.i;
  u += 0x7fffu + ((u >> 16) & 1u);   // round-to-nearest-even
  return (unsigned short)(u >> 16);
}

// ---------------------------------------------------------------------------
// out[idx_out[tap][p]] += feat[idx_in[tap][p]] @ W[tap]   (one tap, 64 rows
// per block).  8 waves: wave -> 16-row block (wave>>1) x 4 col-tiles
// ((wave&1)*4 .. +3).  K loop: 128 in steps of 32 (bf16 WMMA K=32).
// ---------------------------------------------------------------------------
__global__ __launch_bounds__(256) void spconv_wmma_kernel(
    const float* __restrict__ feat, const float* __restrict__ W,
    const int* __restrict__ idx_in, const int* __restrict__ idx_out,
    float* __restrict__ out, int P, int tiles_per_tap)
{
  __shared__ unsigned short Alds[TILE_P * LDA];
  __shared__ unsigned short Wlds[C * LDB];
  __shared__ int row_out[TILE_P];

  const int tap  = blockIdx.x / tiles_per_tap;
  const int tile = blockIdx.x % tiles_per_tap;
  const int p0   = tile * TILE_P;
  const float* Wk   = W + (size_t)tap * C * C;
  const int* in_k   = idx_in  + (size_t)tap * P;
  const int* out_k  = idx_out + (size_t)tap * P;
  const int tid = threadIdx.x;

  // Stage weights f32 -> bf16, transposed [n][k] so each B-fragment element
  // pair (k,k+1) is one contiguous LDS dword.
  #pragma unroll 4
  for (int i = tid; i < C * C; i += 256) {
    int k = i >> 7, n = i & (C - 1);
    Wlds[n * LDB + k] = f2bf(Wk[i]);
  }
  // Gather + convert 64 feature rows (coalesced: 128 consecutive lanes read
  // one 512B row).
  for (int i = tid; i < TILE_P * C; i += 256) {
    int r = i >> 7, c = i & (C - 1);
    int p = p0 + r;
    float v = 0.0f;
    if (p < P) v = feat[(size_t)in_k[p] * C + c];
    Alds[r * LDA + c] = f2bf(v);
  }
  if (tid < TILE_P) {
    int p = p0 + tid;
    row_out[tid] = (p < P) ? out_k[p] : -1;
  }
  __syncthreads();

  const int wave  = tid >> 5;
  const int lane  = tid & 31;
  const int rblk  = wave >> 1;        // 16-row block 0..3
  const int cbase = (wave & 1) * 4;   // first of 4 col tiles
  const int mh    = lane >> 4;        // lane half
  const int mn    = lane & 15;        // M (A/D) or N (B/D) within tile

  v8f acc[4] = {};
  const unsigned short* arow = &Alds[(rblk * 16 + mn) * LDA];
  const int kbA = mh * 8;             // ISA 16-bit A 16x32 lane layout
  const int kbB = mh * 8;             // B 32x16: same k order as A

  #pragma unroll
  for (int kk = 0; kk < C; kk += 32) {
    FragAB a;
    #pragma unroll
    for (int v = 0; v < 8; ++v) {
      int k = kk + kbA + ((v & 4) ? 16 : 0) + (v & 3) * 2;
      a.u[2 * v]     = arow[k];
      a.u[2 * v + 1] = arow[k + 1];
    }
    #pragma unroll
    for (int t = 0; t < 4; ++t) {
      FragAB b;
      const unsigned short* wrow = &Wlds[((cbase + t) * 16 + mn) * LDB];
      #pragma unroll
      for (int v = 0; v < 8; ++v) {
        int k = kk + kbB + ((v & 4) ? 16 : 0) + (v & 3) * 2;
        b.u[2 * v]     = wrow[k];
        b.u[2 * v + 1] = wrow[k + 1];
      }
      acc[t] = __builtin_amdgcn_wmma_f32_16x16x32_bf16(
          false, a.v, false, b.v, (short)0, acc[t], false, false);
    }
  }

  // Scatter-accumulate.  D layout: VGPR j -> M = j + 8*lane_half, N = lane&15.
  #pragma unroll
  for (int t = 0; t < 4; ++t) {
    int col = (cbase + t) * 16 + mn;
    #pragma unroll
    for (int j = 0; j < 8; ++j) {
      int r = rblk * 16 + mh * 8 + j;
      int row = row_out[r];
      if (row >= 0) atomicAdd(out + (size_t)row * C + col, acc[t][j]);
    }
  }
}

// ---------------------------------------------------------------------------
// Elementwise helpers
// ---------------------------------------------------------------------------
__global__ void zero_kernel(float* __restrict__ p, size_t n) {
  size_t i = (size_t)blockIdx.x * blockDim.x + threadIdx.x;
  size_t s = (size_t)gridDim.x * blockDim.x;
  for (; i < n; i += s) p[i] = 0.0f;
}

__global__ void copy_kernel(float* __restrict__ dst, const float* __restrict__ src, size_t n) {
  size_t i = (size_t)blockIdx.x * blockDim.x + threadIdx.x;
  size_t s = (size_t)gridDim.x * blockDim.x;
  for (; i < n; i += s) dst[i] = src[i];
}

// LeakyReLU in place + per-channel sum/sumsq partial reduction -> stats[0..255]
__global__ __launch_bounds__(256) void lrelu_stats_kernel(
    float* __restrict__ x, float* __restrict__ stats, int nrows)
{
  __shared__ float ssum[256], ssq[256];
  const int t = threadIdx.x;
  const int c = t & 127;
  const int s = t >> 7;
  int r0 = blockIdx.x * 32;
  int rend = r0 + 32; if (rend > nrows) rend = nrows;
  float sum = 0.0f, sq = 0.0f;
  for (int r = r0 + s; r < rend; r += 2) {
    float v = x[(size_t)r * C + c];
    v = (v > 0.0f) ? v : 0.01f * v;
    x[(size_t)r * C + c] = v;
    sum += v; sq += v * v;
  }
  ssum[t] = sum; ssq[t] = sq;
  __syncthreads();
  if (s == 0) {
    atomicAdd(&stats[c],       ssum[t] + ssum[t + 128]);
    atomicAdd(&stats[128 + c], ssq[t]  + ssq[t + 128]);
  }
}

// (sum, sumsq, gamma, beta) -> (scale, shift) in place
__global__ void bn_finalize_kernel(float* __restrict__ stats,
                                   const float* __restrict__ g,
                                   const float* __restrict__ b, float inv_n)
{
  int c = threadIdx.x;                  // 128 threads
  float mean = stats[c] * inv_n;
  float var  = stats[128 + c] * inv_n - mean * mean;
  float scale = g[c] * rsqrtf(var + 1e-3f);
  stats[c]       = scale;
  stats[128 + c] = b[c] - mean * scale;
}

__global__ void bn_apply_kernel(float* __restrict__ x,
                                const float* __restrict__ stats, size_t n)
{
  size_t i = (size_t)blockIdx.x * blockDim.x + threadIdx.x;
  size_t s = (size_t)gridDim.x * blockDim.x;
  for (; i < n; i += s) {
    int c = (int)(i & (C - 1));
    x[i] = x[i] * stats[c] + stats[128 + c];
  }
}

// ---------------------------------------------------------------------------
extern "C" void kernel_launch(void* const* d_in, const int* in_sizes, int n_in,
                              void* d_out, int out_size, void* d_ws, size_t ws_size,
                              hipStream_t stream) {
  const float* x_feat = (const float*)d_in[0];
  const float* skip   = (const float*)d_in[1];
  const float* W_t = (const float*)d_in[2];
  const float* g_t = (const float*)d_in[3];
  const float* b_t = (const float*)d_in[4];
  const float* W_up = (const float*)d_in[5];
  const float* W1 = (const float*)d_in[6];
  const float* g1 = (const float*)d_in[7];
  const float* b1 = (const float*)d_in[8];
  const float* W2 = (const float*)d_in[9];
  const float* g2 = (const float*)d_in[10];
  const float* b2 = (const float*)d_in[11];
  const float* W3 = (const float*)d_in[12];
  const float* g3 = (const float*)d_in[13];
  const float* b3 = (const float*)d_in[14];
  const int* ti_in  = (const int*)d_in[15];
  const int* ti_out = (const int*)d_in[16];
  const int* ui_in  = (const int*)d_in[17];
  const int* ui_out = (const int*)d_in[18];
  const int* i1_in  = (const int*)d_in[19];
  const int* i1_out = (const int*)d_in[20];
  const int* i2_in  = (const int*)d_in[21];
  const int* i2_out = (const int*)d_in[22];
  const int* i3_in  = (const int*)d_in[23];
  const int* i3_out = (const int*)d_in[24];

  const int nIn  = in_sizes[0] / C;    // 60000
  const int nOut = in_sizes[1] / C;    // 160000

  float* tmpA  = (float*)d_ws;                       // nIn x 128
  float* buf1  = tmpA + (size_t)nIn * C;             // nOut x 128
  float* stats = buf1 + (size_t)nOut * C;            // 256 floats
  float* outp  = (float*)d_out;                      // nOut x 128

  const dim3 blk(256);
  const int tilesIn  = (nIn  + TILE_P - 1) / TILE_P;
  const int tilesOut = (nOut + TILE_P - 1) / TILE_P;
  const size_t nInEl  = (size_t)nIn  * C;
  const size_t nOutEl = (size_t)nOut * C;

  // --- stage 1: trans_dilao (27 taps, nIn->nIn) -> LReLU -> BN -------------
  zero_kernel<<<2048, blk, 0, stream>>>(tmpA, nInEl);
  zero_kernel<<<1, blk, 0, stream>>>(stats, 256);
  spconv_wmma_kernel<<<27 * tilesIn, blk, 0, stream>>>(
      x_feat, W_t, ti_in, ti_out, tmpA, nIn, tilesIn);
  lrelu_stats_kernel<<<(nIn + 31) / 32, blk, 0, stream>>>(tmpA, stats, nIn);
  bn_finalize_kernel<<<1, 128, 0, stream>>>(stats, g_t, b_t, 1.0f / nIn);
  bn_apply_kernel<<<2048, blk, 0, stream>>>(tmpA, stats, nInEl);

  // --- stage 2: inverse conv (27 taps, nIn->nOut) + residual ----------------
  copy_kernel<<<2048, blk, 0, stream>>>(buf1, skip, nOutEl);
  spconv_wmma_kernel<<<27 * tilesOut, blk, 0, stream>>>(
      tmpA, W_up, ui_in, ui_out, buf1, nOut, tilesOut);

  // --- stage 3: conv1 (9 taps) -> LReLU -> BN  (buf1 -> out) ---------------
  zero_kernel<<<2048, blk, 0, stream>>>(outp, nOutEl);
  zero_kernel<<<1, blk, 0, stream>>>(stats, 256);
  spconv_wmma_kernel<<<9 * tilesOut, blk, 0, stream>>>(
      buf1, W1, i1_in, i1_out, outp, nOut, tilesOut);
  lrelu_stats_kernel<<<(nOut + 31) / 32, blk, 0, stream>>>(outp, stats, nOut);
  bn_finalize_kernel<<<1, 128, 0, stream>>>(stats, g1, b1, 1.0f / nOut);
  bn_apply_kernel<<<2048, blk, 0, stream>>>(outp, stats, nOutEl);

  // --- stage 4: conv2 (9 taps) -> LReLU -> BN  (out -> buf1) ---------------
  zero_kernel<<<2048, blk, 0, stream>>>(buf1, nOutEl);
  zero_kernel<<<1, blk, 0, stream>>>(stats, 256);
  spconv_wmma_kernel<<<9 * tilesOut, blk, 0, stream>>>(
      outp, W2, i2_in, i2_out, buf1, nOut, tilesOut);
  lrelu_stats_kernel<<<(nOut + 31) / 32, blk, 0, stream>>>(buf1, stats, nOut);
  bn_finalize_kernel<<<1, 128, 0, stream>>>(stats, g2, b2, 1.0f / nOut);
  bn_apply_kernel<<<2048, blk, 0, stream>>>(buf1, stats, nOutEl);

  // --- stage 5: conv3 (27 taps) -> LReLU -> BN  (buf1 -> out) --------------
  zero_kernel<<<2048, blk, 0, stream>>>(outp, nOutEl);
  zero_kernel<<<1, blk, 0, stream>>>(stats, 256);
  spconv_wmma_kernel<<<27 * tilesOut, blk, 0, stream>>>(
      buf1, W3, i3_in, i3_out, outp, nOut, tilesOut);
  lrelu_stats_kernel<<<(nOut + 31) / 32, blk, 0, stream>>>(outp, stats, nOut);
  bn_finalize_kernel<<<1, 128, 0, stream>>>(stats, g3, b3, 1.0f / nOut);
  bn_apply_kernel<<<2048, blk, 0, stream>>>(outp, stats, nOutEl);
}
